// BlockSoftmaxLinearHybrid_16226386444500
// MI455X (gfx1250) — matrix operand order repaired and checked by reading, not compiled, not run on hardware
//
#include <hip/hip_runtime.h>

// ---------------------------------------------------------------------------
// Hedgehog hybrid block-softmax / linear attention for gfx1250 (MI455X).
// One workgroup (8 wave32) per (b,h); sequential scan over 64 key blocks.
//  - All GEMMs on v_wmma_f32_16x16x32_f16 (fp32 accumulate).
//  - q/k/v staged one block ahead with GLOBAL_LOAD_ASYNC_TO_LDS_B128
//    (double-buffered fp32 staging in LDS, s_wait_asynccnt pipeline).
//  - Row softmax/exp/denominator passes: 4 lanes/row + __shfl_xor reductions.
//  - fp32 recurrent state (128x64) lives in LDS, accumulated via WMMA C.
// ---------------------------------------------------------------------------

typedef __attribute__((ext_vector_type(16))) _Float16 v16h;
typedef __attribute__((ext_vector_type(8)))  _Float16 v8h;
typedef __attribute__((ext_vector_type(8)))  float    v8f;

namespace {

constexpr int Hh   = 16;    // heads
constexpr int L    = 4096;  // sequence length
constexpr int D    = 64;    // head dim
constexpr int Fdim = 64;    // feature dim
constexpr int S    = 64;    // block size
constexpr int NB   = L / S; // 64 blocks
constexpr int F2   = 2 * Fdim;

constexpr int PH  = 72;     // f16 pitch for 64-wide tiles   (144 B rows)
constexpr int PH2 = 136;    // f16 pitch for 128-wide tiles  (272 B rows)
constexpr int PF  = 65;     // f32 pitch for 64-wide tiles

// LDS layout (all offsets multiples of 16 bytes)
constexpr int OFF_Q   = 0;
constexpr int OFF_K   = OFF_Q   + S    * PH  * 2;   // q block  (s,d) f16
constexpr int OFF_VT  = OFF_K   + S    * PH  * 2;   // k block  (t,d) f16
constexpr int OFF_WT  = OFF_VT  + D    * PH  * 2;   // v^T      (d,t) f16
constexpr int OFF_PQ  = OFF_WT  + Fdim * PH  * 2;   // W^T      (f,d) f16
constexpr int OFF_PKT = OFF_PQ  + S    * PH2 * 2;   // phi_q    (s,f2) f16
constexpr int OFF_STH = OFF_PKT + F2   * PH  * 2;   // phi_k^T  (f2,s) f16
constexpr int OFF_AH  = OFF_STH + D    * PH2 * 2;   // state^T  (d,f2) f16
constexpr int OFF_U   = OFF_AH  + S    * PH  * 2;   // exp(scores) (s,t) f16
constexpr int OFF_S   = OFF_U   + S    * PF  * 4;   // scratch u/scores/out f32
constexpr int OFF_Z   = OFF_S   + F2   * PF  * 4;   // state  (f2,d) f32
constexpr int OFF_DEN = OFF_Z   + F2   * 4;         // Z (f2) f32
constexpr int OFF_STG = OFF_DEN + S * 4;            // 1/denominator per row
// Double-buffered raw fp32 staging for async q/k/v copies.
constexpr int STG_T   = S * D * 4;                  // 16 KB per tensor
constexpr int STG_BUF = 3 * STG_T;                  // 48 KB per buffer
constexpr int SMEM_BYTES = OFF_STG + 2 * STG_BUF;   // ~243 KB (< 320 KB/WGP)

union V16U { v16h v; v8h h[2]; };

// A fragment: row-major (M,K) source. Lane layout per CDNA5 ISA 7.12.2.
__device__ __forceinline__ v16h fragA(const _Float16* s, int pitch, int mbase,
                                      int kbase, int l16, int half) {
  const _Float16* p = s + (mbase + l16) * pitch + kbase + half * 8;
  V16U u;
  u.h[0] = *(const v8h*)(p);
  u.h[1] = *(const v8h*)(p + 16);
  return u.v;
}

// B fragment from B^T stored row-major (N,K).
__device__ __forceinline__ v16h fragB(const _Float16* s, int pitch, int nbase,
                                      int kbase, int l16, int half) {
  const _Float16* p = s + (nbase + l16) * pitch + kbase + half * 8;
  V16U u;
  u.h[0] = *(const v8h*)(p);
  u.h[1] = *(const v8h*)(p + 16);
  return u.v;
}

__device__ __forceinline__ void storeC(float* d, int pitch, int mbase, int nbase,
                                       int l16, int half, v8f c) {
  float* p = d + (mbase + half * 8) * pitch + nbase + l16;
#pragma unroll
  for (int r = 0; r < 8; ++r) p[r * pitch] = c[r];
}

__device__ __forceinline__ v8f loadC(const float* d, int pitch, int mbase, int nbase,
                                     int l16, int half) {
  const float* p = d + (mbase + half * 8) * pitch + nbase + l16;
  v8f c;
#pragma unroll
  for (int r = 0; r < 8; ++r) c[r] = p[r * pitch];
  return c;
}

__device__ __forceinline__ v8f wmma16(v16h a, v16h b, v8f c) {
  return __builtin_amdgcn_wmma_f32_16x16x32_f16(false, a, false, b, (short)0, c,
                                                false, false);
}

// Async global->LDS copy, saddr form: LDS address comes from the vdst VGPR plus the instruction offset,
// gaddr = saddr64 + voffset32.  Tracked by ASYNCcnt.
__device__ __forceinline__ void async_to_lds_b128(unsigned lds_off, const void* sbase,
                                                  unsigned voff) {
  asm volatile("global_load_async_to_lds_b128 %0, %1, %2"
               :: "v"(lds_off), "v"(voff), "s"(sbase)
               : "memory");
}

// Issue one block's q/k/v (3 x 16 KB) into staging buffer `buf`: 12 async ops/wave.
__device__ __forceinline__ void issue_block_async(const float* q, const float* k,
                                                  const float* v, size_t blk,
                                                  int buf, int tid) {
  const unsigned base = (unsigned)(OFF_STG + buf * STG_BUF) + tid * 16;
  const unsigned voff = tid * 16;  // bytes within the 16 KB tensor block
#pragma unroll
  for (int j = 0; j < 4; ++j) {
    async_to_lds_b128(base + 0 * STG_T + j * 4096, q + blk, voff + j * 4096);
    async_to_lds_b128(base + 1 * STG_T + j * 4096, k + blk, voff + j * 4096);
    async_to_lds_b128(base + 2 * STG_T + j * 4096, v + blk, voff + j * 4096);
  }
}

} // namespace

__global__ __launch_bounds__(256)
void hedgehog_hybrid_kernel(const float* __restrict__ q, const float* __restrict__ k,
                            const float* __restrict__ v, const float* __restrict__ W,
                            const float* __restrict__ alpha, float* __restrict__ out) {
  extern __shared__ char smem[];
  _Float16* sQ   = (_Float16*)(smem + OFF_Q);
  _Float16* sK   = (_Float16*)(smem + OFF_K);
  _Float16* sVT  = (_Float16*)(smem + OFF_VT);
  _Float16* sWT  = (_Float16*)(smem + OFF_WT);
  _Float16* sPQ  = (_Float16*)(smem + OFF_PQ);
  _Float16* sPKT = (_Float16*)(smem + OFF_PKT);
  _Float16* sSTH = (_Float16*)(smem + OFF_STH);
  _Float16* sAH  = (_Float16*)(smem + OFF_AH);
  float*    sU   = (float*)(smem + OFF_U);   // u / scores / output staging
  float*    sS   = (float*)(smem + OFF_S);   // fp32 state (f2,d)
  float*    sZ   = (float*)(smem + OFF_Z);
  float*    sDen = (float*)(smem + OFF_DEN);

  const int tid  = threadIdx.x;
  const int lane = tid & 31;
  const int l16  = lane & 15;
  const int half = lane >> 4;
  const int wave = tid >> 5;
  const int row  = tid >> 2;       // row-pass mapping: 4 lanes per row
  const int q4   = tid & 3;        // 16 columns per lane

  const int wg = blockIdx.x;       // b*H + h
  const int h  = wg & (Hh - 1);

  const float wsig    = 1.0f / (1.0f + __expf(-alpha[h]));
  const float scaling = 0.125f;    // D^-0.5
  const size_t bh_off = (size_t)wg * (size_t)(L * D);

  // ---- kick off async staging of block 0 ----
  issue_block_async(q, k, v, bh_off, 0, tid);

  // ---- one-time init: W^T tile, zero state ----
  for (int i = tid; i < D * Fdim; i += 256) {
    int d = i >> 6, f = i & 63;
    sWT[f * PH + d] = (_Float16)W[(size_t)h * D * Fdim + i];  // W[h][d][f]
  }
  for (int i = tid; i < F2 * PF; i += 256) sS[i] = 0.0f;
  for (int i = tid; i < D * PH2; i += 256) sSTH[i] = (_Float16)0.0f;
  if (tid < F2) sZ[tid] = 0.0f;
  __syncthreads();

  for (int n = 0; n < NB; ++n) {
    const size_t blk = bh_off + (size_t)n * (S * D);
    const int    buf = n & 1;

    // ---- prefetch block n+1 asynchronously; wait for block n's data ----
    if (n + 1 < NB) {
      issue_block_async(q, k, v, blk + (size_t)(S * D), buf ^ 1, tid);
      asm volatile("s_wait_asynccnt 0xc" ::: "memory");  // 12 in flight => block n done
    } else {
      asm volatile("s_wait_asynccnt 0x0" ::: "memory");
    }
    __syncthreads();   // all waves waited -> staging buffer `buf` fully visible

    // ---- convert staged fp32 -> f16 tiles (v transposed) ----
    {
      const float* stq = (const float*)(smem + OFF_STG + buf * STG_BUF);
      const float* stk = stq + S * D;
      const float* stv = stk + S * D;
      for (int i = tid; i < S * D; i += 256) {
        int s = i >> 6, d = i & 63;
        sQ[s * PH + d]  = (_Float16)stq[i];
        sK[s * PH + d]  = (_Float16)stk[i];
        sVT[d * PH + s] = (_Float16)stv[i];
      }
    }
    __syncthreads();

    // ---- u_q = q · W  (16 tiles, 2 per wave) ----
#pragma unroll
    for (int i = 0; i < 2; ++i) {
      int T = wave * 2 + i, mt = (T >> 2) * 16, nt = (T & 3) * 16;
      v8f c = {};
      c = wmma16(fragA(sQ, PH, mt, 0,  l16, half), fragB(sWT, PH, nt, 0,  l16, half), c);
      c = wmma16(fragA(sQ, PH, mt, 32, l16, half), fragB(sWT, PH, nt, 32, l16, half), c);
      storeC(sU, PF, mt, nt, l16, half, c);
    }
    __syncthreads();

    // ---- phi_q = [softmax(u), softmax(-u)]; 4 lanes/row, shfl reductions ----
    {
      const float* ur = sU + row * PF + q4 * 16;
      float mx = -3.0e38f, mn = 3.0e38f;
#pragma unroll
      for (int f = 0; f < 16; ++f) { float uv = ur[f]; mx = fmaxf(mx, uv); mn = fminf(mn, uv); }
      mx = fmaxf(mx, __shfl_xor(mx, 1)); mx = fmaxf(mx, __shfl_xor(mx, 2));
      mn = fminf(mn, __shfl_xor(mn, 1)); mn = fminf(mn, __shfl_xor(mn, 2));
      float e1[16], e2[16], s1 = 0.f, s2 = 0.f;
#pragma unroll
      for (int f = 0; f < 16; ++f) {
        float uv = ur[f];
        e1[f] = __expf(uv - mx); e2[f] = __expf(mn - uv);
        s1 += e1[f]; s2 += e2[f];
      }
      s1 += __shfl_xor(s1, 1); s1 += __shfl_xor(s1, 2);
      s2 += __shfl_xor(s2, 1); s2 += __shfl_xor(s2, 2);
      float i1 = 1.f / s1, i2 = 1.f / s2;
      _Float16* pr = sPQ + row * PH2 + q4 * 16;
#pragma unroll
      for (int f = 0; f < 16; ++f) {
        pr[f]        = (_Float16)(e1[f] * i1);
        pr[Fdim + f] = (_Float16)(e2[f] * i2);
      }
    }
    __syncthreads();

    // ---- u_k = k · W ----
#pragma unroll
    for (int i = 0; i < 2; ++i) {
      int T = wave * 2 + i, mt = (T >> 2) * 16, nt = (T & 3) * 16;
      v8f c = {};
      c = wmma16(fragA(sK, PH, mt, 0,  l16, half), fragB(sWT, PH, nt, 0,  l16, half), c);
      c = wmma16(fragA(sK, PH, mt, 32, l16, half), fragB(sWT, PH, nt, 32, l16, half), c);
      storeC(sU, PF, mt, nt, l16, half, c);
    }
    __syncthreads();

    // ---- phi_k, written transposed (f2, s) for the state-update WMMA ----
    {
      const float* ur = sU + row * PF + q4 * 16;
      float mx = -3.0e38f, mn = 3.0e38f;
#pragma unroll
      for (int f = 0; f < 16; ++f) { float uv = ur[f]; mx = fmaxf(mx, uv); mn = fminf(mn, uv); }
      mx = fmaxf(mx, __shfl_xor(mx, 1)); mx = fmaxf(mx, __shfl_xor(mx, 2));
      mn = fminf(mn, __shfl_xor(mn, 1)); mn = fminf(mn, __shfl_xor(mn, 2));
      float e1[16], e2[16], s1 = 0.f, s2 = 0.f;
#pragma unroll
      for (int f = 0; f < 16; ++f) {
        float uv = ur[f];
        e1[f] = __expf(uv - mx); e2[f] = __expf(mn - uv);
        s1 += e1[f]; s2 += e2[f];
      }
      s1 += __shfl_xor(s1, 1); s1 += __shfl_xor(s1, 2);
      s2 += __shfl_xor(s2, 1); s2 += __shfl_xor(s2, 2);
      float i1 = 1.f / s1, i2 = 1.f / s2;
#pragma unroll
      for (int f = 0; f < 16; ++f) {
        int fg = q4 * 16 + f;
        sPKT[fg * PH + row]          = (_Float16)(e1[f] * i1);
        sPKT[(Fdim + fg) * PH + row] = (_Float16)(e2[f] * i2);
      }
    }
    __syncthreads();

    // ---- scores = q · k^T (raw; scale folded into exp) ----
#pragma unroll
    for (int i = 0; i < 2; ++i) {
      int T = wave * 2 + i, mt = (T >> 2) * 16, nt = (T & 3) * 16;
      v8f c = {};
      c = wmma16(fragA(sQ, PH, mt, 0,  l16, half), fragB(sK, PH, nt, 0,  l16, half), c);
      c = wmma16(fragA(sQ, PH, mt, 32, l16, half), fragB(sK, PH, nt, 32, l16, half), c);
      storeC(sU, PF, mt, nt, l16, half, c);
    }
    __syncthreads();

    // ---- a = exp(scores*scale - max), sm_den, lin_den, 1/denominator ----
    {
      const float* sr = sU + row * PF + q4 * 16;
      float mx = -3.0e38f;
#pragma unroll
      for (int t = 0; t < 16; ++t) mx = fmaxf(mx, sr[t]);
      mx = fmaxf(mx, __shfl_xor(mx, 1)); mx = fmaxf(mx, __shfl_xor(mx, 2));
      float ssum = 0.f;
      _Float16* ar = sAH + row * PH + q4 * 16;
#pragma unroll
      for (int t = 0; t < 16; ++t) {
        float e = __expf((sr[t] - mx) * scaling);
        ar[t] = (_Float16)e;
        ssum += e;
      }
      ssum += __shfl_xor(ssum, 1); ssum += __shfl_xor(ssum, 2);
      // lin_den = phi_q . Z (pre-update), 32 features per lane
      const _Float16* pr = sPQ + row * PH2 + q4 * 32;
      const float*    zr = sZ + q4 * 32;
      float dot = 0.f;
#pragma unroll
      for (int f = 0; f < 32; ++f) dot += (float)pr[f] * zr[f];
      dot += __shfl_xor(dot, 1); dot += __shfl_xor(dot, 2);
      if (q4 == 0) {
        float smden = fmaxf(ssum, 1e-6f);
        float lden  = fmaxf(dot, 1e-6f);
        sDen[row] = 1.0f / fmaxf(wsig * smden + lden, 1e-6f);
      }
    }
    __syncthreads();

    // ---- sm_num = a·v, lin_num = phi_q·S_state (pre-update), combine ----
#pragma unroll
    for (int i = 0; i < 2; ++i) {
      int T = wave * 2 + i, mt = (T >> 2) * 16, nt = (T & 3) * 16;
      v8f csm = {};
      csm = wmma16(fragA(sAH, PH, mt, 0,  l16, half), fragB(sVT, PH, nt, 0,  l16, half), csm);
      csm = wmma16(fragA(sAH, PH, mt, 32, l16, half), fragB(sVT, PH, nt, 32, l16, half), csm);
      v8f clin = {};
#pragma unroll
      for (int kc = 0; kc < F2; kc += 32)
        clin = wmma16(fragA(sPQ, PH2, mt, kc, l16, half), fragB(sSTH, PH2, nt, kc, l16, half), clin);
      float* op = sU + (mt + half * 8) * PF + nt + l16;
      int mrow = mt + half * 8;
#pragma unroll
      for (int r = 0; r < 8; ++r)
        op[r * PF] = (wsig * csm[r] + clin[r]) * sDen[mrow + r];
    }
    __syncthreads();

    // ---- coalesced output store (block is 4096 contiguous floats) ----
    for (int i = tid; i < S * D; i += 256)
      out[blk + i] = sU[(i >> 6) * PF + (i & 63)];

    // ---- state update: S += phi_k^T · v   (128x64, fp32 accumulate in LDS) ----
#pragma unroll
    for (int j = 0; j < 4; ++j) {
      int mt = wave * 16, nt = j * 16;
      v8f c = loadC(sS, PF, mt, nt, l16, half);
      c = wmma16(fragA(sPKT, PH, mt, 0,  l16, half), fragB(sVT, PH, nt, 0,  l16, half), c);
      c = wmma16(fragA(sPKT, PH, mt, 32, l16, half), fragB(sVT, PH, nt, 32, l16, half), c);
      storeC(sS, PF, mt, nt, l16, half, c);
    }
    // ---- Z += colsum(phi_k): 2 lanes per feature ----
    {
      int f = tid >> 1, part = tid & 1;
      const _Float16* kr = sPKT + f * PH + part * 32;
      float z = 0.f;
#pragma unroll
      for (int s2 = 0; s2 < 32; ++s2) z += (float)kr[s2];
      z += __shfl_xor(z, 1);
      if (part == 0) sZ[f] += z;
    }
    __syncthreads();

    // ---- rebuild f16 transposed state (d, f2) for next block's lin_num ----
    for (int i = tid; i < F2 * D; i += 256) {
      int f = i >> 6, d = i & 63;
      sSTH[d * PH2 + f] = (_Float16)sS[f * PF + d];
    }
    __syncthreads();
  }
}

extern "C" void kernel_launch(void* const* d_in, const int* in_sizes, int n_in,
                              void* d_out, int out_size, void* d_ws, size_t ws_size,
                              hipStream_t stream) {
  (void)n_in; (void)out_size; (void)d_ws; (void)ws_size;
  const float* q     = (const float*)d_in[0];
  const float* k     = (const float*)d_in[1];
  const float* v     = (const float*)d_in[2];
  const float* W     = (const float*)d_in[3];
  const float* alpha = (const float*)d_in[4];
  float* out = (float*)d_out;

  const int BH = in_sizes[0] / (L * D);  // B*H = 32 workgroups
  hedgehog_hybrid_kernel<<<dim3(BH), dim3(256), SMEM_BYTES, stream>>>(q, k, v, W, alpha, out);
}
